// ConformerBlock_61383672594466
// MI455X (gfx1250) — matrix operand order repaired and checked by reading, not compiled, not run on hardware
//
#include <hip/hip_runtime.h>
#include <hip/hip_bf16.h>
#include <cstdint>

// ---------------------------------------------------------------------------
// Types / helpers
// ---------------------------------------------------------------------------
typedef __attribute__((ext_vector_type(16))) __bf16 bf16x16;
typedef __attribute__((ext_vector_type(8)))  float  f32x8;

struct alignas(16) U4 { unsigned int x, y, z, w; };
union Frag  { U4 q[2]; bf16x16 v; };
union U4H8  { U4 q; unsigned short s[8]; };

static __device__ __forceinline__ unsigned short f2bf(float f) {
  union { float f; unsigned u; } x; x.f = f;
  unsigned u = x.u;
  unsigned r = (u + 0x7FFFu + ((u >> 16) & 1u)) >> 16;  // round-to-nearest-even
  return (unsigned short)r;
}
static __device__ __forceinline__ float sigmoidf_(float x) { return 1.0f / (1.0f + __expf(-x)); }
static __device__ __forceinline__ float siluf_(float x)    { return x * sigmoidf_(x); }

// CDNA5 async global->LDS DMA (16B per lane), tracked by ASYNCcnt.
// LDS destination = low 32 bits of the generic shared-memory address
// (ISA: LDS aperture addresses truncate to LDS offset).
static __device__ __forceinline__ void async_lds_b128(void* lds_ptr, const void* gptr) {
  unsigned lo = (unsigned)(uintptr_t)lds_ptr;
  asm volatile("global_load_async_to_lds_b128 %0, %1, off"
               :: "v"(lo), "v"(gptr) : "memory");
}
static __device__ __forceinline__ void wait_async0() {
  asm volatile("s_wait_asynccnt 0" ::: "memory");
}

// ---------------------------------------------------------------------------
// Elementwise kernels
// ---------------------------------------------------------------------------
__global__ void k_f32_to_bf16(const float* __restrict__ src, unsigned short* __restrict__ dst, long n) {
  long i = (long)blockIdx.x * blockDim.x + threadIdx.x;
  if (i < n) dst[i] = f2bf(src[i]);
}

// dconv_w [O,I,1,K] (fp32) -> [K][O][I] (bf16): each tap becomes a contiguous row-major matrix
__global__ void k_reorder_dconv(const float* __restrict__ src, unsigned short* __restrict__ dst,
                                int O, int I, int K) {
  long n = (long)O * I * K;
  long i = (long)blockIdx.x * blockDim.x + threadIdx.x;
  if (i >= n) return;
  int k = (int)(i % K);
  long oi = i / K;
  int ii = (int)(oi % I);
  int o  = (int)(oi / I);
  dst[(long)k * O * I + (long)o * I + ii] = f2bf(src[i]);
}

// y = x*g[c]+b[c], bf16 out.  x: [B,C,S] flat
__global__ void k_bn_bf16(const float* __restrict__ x, const float* __restrict__ g,
                          const float* __restrict__ b, unsigned short* __restrict__ dst,
                          long n, int C, int S) {
  long i = (long)blockIdx.x * blockDim.x + threadIdx.x;
  if (i >= n) return;
  int c = (int)((i / S) % C);
  dst[i] = f2bf(fmaf(x[i], g[c], b[c]));
}

// final BN, fp32 out
__global__ void k_bn_f32(const float* __restrict__ x, const float* __restrict__ g,
                         const float* __restrict__ b, float* __restrict__ dst,
                         long n, int C, int S) {
  long i = (long)blockIdx.x * blockDim.x + threadIdx.x;
  if (i >= n) return;
  int c = (int)((i / S) % C);
  dst[i] = fmaf(x[i], g[c], b[c]);
}

// GLU: h [B, 2*Cin, S] fp32 -> dst [B, Cin, S] bf16
__global__ void k_glu_bf16(const float* __restrict__ h, unsigned short* __restrict__ dst,
                           int Cin, int S, int B) {
  long n = (long)B * Cin * S;
  long i = (long)blockIdx.x * blockDim.x + threadIdx.x;
  if (i >= n) return;
  int s = (int)(i % S);
  long t = i / S;
  int c = (int)(t % Cin);
  int b = (int)(t / Cin);
  float a  = h[((long)b * 2 * Cin + c) * S + s];
  float gg = h[((long)b * 2 * Cin + Cin + c) * S + s];
  dst[i] = f2bf(a * sigmoidf_(gg));
}

// per-z transpose + convert: src [R,C] fp32 -> dst [C,R] bf16, z = blockIdx.z
__global__ void k_transpose_cvt(const float* __restrict__ src, unsigned short* __restrict__ dst,
                                int R, int C) {
  long rc = (long)R * C;
  long i = (long)blockIdx.x * blockDim.x + threadIdx.x;
  if (i >= rc) return;
  int c = (int)(i % C);
  int r = (int)(i / C);
  long base = (long)blockIdx.z * rc;
  dst[base + (long)c * R + r] = f2bf(src[base + (long)r * C + c]);
}

// row softmax, fp32 in -> bf16 out. one block (128 thr) per row of length S
__global__ void k_softmax_bf16(const float* __restrict__ scores, unsigned short* __restrict__ w, int S) {
  long base = (long)blockIdx.x * S;
  __shared__ float red[128];
  float mx = -1e30f;
  for (int c = threadIdx.x; c < S; c += 128) mx = fmaxf(mx, scores[base + c]);
  red[threadIdx.x] = mx; __syncthreads();
  for (int o = 64; o > 0; o >>= 1) {
    if (threadIdx.x < o) red[threadIdx.x] = fmaxf(red[threadIdx.x], red[threadIdx.x + o]);
    __syncthreads();
  }
  mx = red[0]; __syncthreads();
  float sm = 0.f;
  for (int c = threadIdx.x; c < S; c += 128) sm += __expf(scores[base + c] - mx);
  red[threadIdx.x] = sm; __syncthreads();
  for (int o = 64; o > 0; o >>= 1) {
    if (threadIdx.x < o) red[threadIdx.x] += red[threadIdx.x + o];
    __syncthreads();
  }
  float inv = 1.0f / red[0];
  for (int c = threadIdx.x; c < S; c += 128)
    w[base + c] = f2bf(__expf(scores[base + c] - mx) * inv);
}

// ---------------------------------------------------------------------------
// Generalized bf16 WMMA GEMM (+ multi-tap conv via halo-staged B, fused epilogue)
//   C[z][m,n] = epi( sum_t sum_k A[z][t][m,k] * B[z][k, n + t - pad] )
//   epi: +bias[m]; optional *bng[m]+bnb[m]; optional SiLU; *alpha; +resid
// Requirements: Kd % 32 == 0; lda % 8 == 0; M % 16 == 0 (OOB rows clamped,
// results masked in epilogue).  CONV=true: B tile staged once per K-chunk with
// a 160-col halo (all 31 taps reuse it from LDS); A tile is double-buffered
// and fetched with async global->LDS DMA overlapping the previous tap's WMMAs.
// ---------------------------------------------------------------------------
struct GemmP {
  const unsigned short* A; long sAz; int lda; long tapStrideA; int taps; int pad;
  const unsigned short* Bm; long sBz; int ldb; int Nsrc;
  float* Cf; unsigned short* Cb; long sCz; int ldc;
  const float* bias; const float* bng; const float* bnb;
  const float* resid; long sRz;
  float alpha; int M; int N; int Kd; int act;  // act: 0=none, 1=SiLU
};

#define BM 64
#define BN 128
#define BK 32

template <bool CONV>
__launch_bounds__(256)
__global__ void k_gemm_wmma(GemmP p) {
  constexpr int HC   = CONV ? (BN + 32) : BN;  // staged B columns (halo for taps)
  constexpr int CPR  = HC / 8;                 // 8-col chunks per k-row
  constexpr int OFFB = CONV ? 16 : 0;          // halo offset of col 0
  constexpr int NAB  = CONV ? 2 : 1;           // A tile buffers (double-buffer taps)

  __shared__ unsigned short As[NAB][BM][BK];   // A tile(s), row-major [m][k]
  __shared__ unsigned short Bs[HC][BK];        // B tile, N-major [n][k]

  const int tid  = threadIdx.x;
  const int lane = tid & 31;
  const int wave = tid >> 5;        // 0..7
  const int wm   = wave & 3;        // M sub-tile (16 rows each)
  const int wn   = wave >> 2;       // N half (64 cols each)
  const int z    = blockIdx.z;
  const int m0   = blockIdx.y * BM;
  const int n0   = blockIdx.x * BN;

  const unsigned short* Az = p.A  + (long)z * p.sAz;
  const unsigned short* Bz = p.Bm + (long)z * p.sBz;

  f32x8 acc[4];
#pragma unroll
  for (int j = 0; j < 4; ++j)
#pragma unroll
    for (int r = 0; r < 8; ++r) acc[j][r] = 0.0f;

  // ISA 16-bit A layout: lanes 0-15 -> K {0..7,16..23}; lanes 16-31 -> K {8..15,24..31}
  const int rowA = lane & 15;
  const int kgA  = (lane < 16) ? 0 : 8;
  // B layout: col across lanes; lanes 0-15 -> K {0..7,16..23}; lanes 16-31 -> K {8..15,24..31}
  const int kr0B = (lane < 16) ? 0 : 8;

  // A tile stage via async DMA: one 16B chunk per thread, row clamped
  auto stageA = [&](const unsigned short* At, int kc, unsigned short (*Ab)[BK]) {
    int r = tid >> 2;                 // 0..63
    int c = (tid & 3) * 8;            // 0,8,16,24
    int gm = m0 + r; gm = (gm < p.M) ? gm : (p.M - 1);
    async_lds_b128(&Ab[r][c], At + (long)gm * p.lda + kc + c);
  };

  // B halo stage (transposed into Bs[n][k]); block-uniform fast path when the
  // whole staged window is in bounds (base col is 8-aligned by construction)
  auto stageB = [&](int kc) {
    const int c0 = n0 - OFFB;
    if (c0 >= 0 && c0 + HC <= p.Nsrc) {          // uniform across block
#pragma unroll
      for (int rep = 0; rep < (32 * CPR + 255) / 256; ++rep) {
        int cid = tid + rep * 256;
        if (32 * CPR % 256 != 0 && cid >= 32 * CPR) break;
        int k  = cid / CPR;
        int nc = (cid % CPR) * 8;
        const unsigned short* brow = Bz + (long)(kc + k) * p.ldb;
        U4H8 u;
        u.q = *(const U4*)(brow + c0 + nc);
#pragma unroll
        for (int e = 0; e < 8; ++e) Bs[nc + e][k] = u.s[e];
      }
    } else {
#pragma unroll
      for (int rep = 0; rep < (32 * CPR + 255) / 256; ++rep) {
        int cid = tid + rep * 256;
        if (32 * CPR % 256 != 0 && cid >= 32 * CPR) break;
        int k  = cid / CPR;
        int nc = (cid % CPR) * 8;
        const unsigned short* brow = Bz + (long)(kc + k) * p.ldb;
        U4H8 u;
#pragma unroll
        for (int e = 0; e < 8; ++e) {
          int c = c0 + nc + e;
          int cc = c < 0 ? 0 : (c >= p.Nsrc ? p.Nsrc - 1 : c);
          unsigned short lv = brow[cc];
          u.s[e] = ((unsigned)c < (unsigned)p.Nsrc) ? lv : (unsigned short)0;
        }
#pragma unroll
        for (int e = 0; e < 8; ++e) Bs[nc + e][k] = u.s[e];
      }
    }
  };

  // per-tap compute: A frag + 4 B frags from LDS, 4 WMMAs
  auto compute = [&](const unsigned short (*Ab)[BK], int shift) {
    Frag fa;
    fa.q[0] = *(const U4*)&Ab[wm * 16 + rowA][kgA];
    fa.q[1] = *(const U4*)&Ab[wm * 16 + rowA][16 + kgA];
    Frag fb[4];
#pragma unroll
    for (int j = 0; j < 4; ++j) {
      int col = wn * 64 + j * 16 + (lane & 15) + shift + OFFB;
      fb[j].q[0] = *(const U4*)&Bs[col][kr0B];
      fb[j].q[1] = *(const U4*)&Bs[col][16 + kr0B];
    }
#pragma unroll
    for (int j = 0; j < 4; ++j)
      acc[j] = __builtin_amdgcn_wmma_f32_16x16x32_bf16(
          false, fa.v, false, fb[j].v, (short)0, acc[j], false, false);
  };

  if (!CONV) {
    for (int kc = 0; kc < p.Kd; kc += BK) {
      __syncthreads();               // previous-iteration LDS readers done
      stageA(Az, kc, As[0]);         // async DMA overlaps B staging below
      stageB(kc);
      wait_async0();
      __syncthreads();
      compute(As[0], 0);
    }
  } else {
    for (int kc = 0; kc < p.Kd; kc += BK) {
      if (kc == 0) __syncthreads();  // later iterations covered by tap-loop barrier
      stageB(kc);
      stageA(Az, kc, As[0]);         // tap 0 prologue
      wait_async0();
      __syncthreads();
      for (int t = 0; t < p.taps; ++t) {
        if (t + 1 < p.taps)          // prefetch next tap's weights via async DMA
          stageA(Az + (long)(t + 1) * p.tapStrideA, kc, As[(t + 1) & 1]);
        compute(As[t & 1], t - p.pad);   // overlaps with in-flight DMA
        wait_async0();
        __syncthreads();
      }
    }
  }

  // --- fused epilogue ---
  const int colBase = n0 + wn * 64 + (lane & 15);
#pragma unroll
  for (int j = 0; j < 4; ++j) {
    int col = colBase + j * 16;
    if (col >= p.N) continue;
#pragma unroll
    for (int r = 0; r < 8; ++r) {
      int row = m0 + wm * 16 + ((lane < 16) ? r : 8 + r);
      if (row >= p.M) continue;
      float v = acc[j][r];
      if (p.bias) v += p.bias[row];
      if (p.bng)  v = fmaf(v, p.bng[row], p.bnb[row]);
      if (p.act == 1) v = siluf_(v);
      v *= p.alpha;
      long ci = (long)z * p.sCz + (long)row * p.ldc + col;
      if (p.resid) v += p.resid[(long)z * p.sRz + (long)row * p.ldc + col];
      if (p.Cf) p.Cf[ci] = v;
      if (p.Cb) p.Cb[ci] = f2bf(v);
    }
  }
}

// ---------------------------------------------------------------------------
// Host orchestration
// ---------------------------------------------------------------------------
extern "C" void kernel_launch(void* const* d_in, const int* in_sizes, int n_in,
                              void* d_out, int out_size, void* d_ws, size_t ws_size,
                              hipStream_t stream) {
  (void)in_sizes; (void)n_in; (void)out_size; (void)ws_size;
  const int B = 8, DIM = 512, S = 1024, H = 8, DH = 64, FF = 1024, CI = 1024, KW = 31, PADC = 15;

  const float* x      = (const float*)d_in[0];
  const float* ff1_g  = (const float*)d_in[1];
  const float* ff1_b  = (const float*)d_in[2];
  const float* ff1_w1 = (const float*)d_in[3];
  const float* ff1_b1 = (const float*)d_in[4];
  const float* ff1_w2 = (const float*)d_in[5];
  const float* ff1_b2 = (const float*)d_in[6];
  const float* attn_g = (const float*)d_in[7];
  const float* attn_b = (const float*)d_in[8];
  const float* wq     = (const float*)d_in[9];
  const float* bq     = (const float*)d_in[10];
  const float* wk     = (const float*)d_in[11];
  const float* bk     = (const float*)d_in[12];
  const float* wv     = (const float*)d_in[13];
  const float* bv     = (const float*)d_in[14];
  const float* wo     = (const float*)d_in[15];
  const float* bo     = (const float*)d_in[16];
  const float* conv_g = (const float*)d_in[17];
  const float* conv_b = (const float*)d_in[18];
  const float* pw1_w  = (const float*)d_in[19];
  const float* pw1_b  = (const float*)d_in[20];
  const float* dconv_w= (const float*)d_in[21];
  const float* dconv_b= (const float*)d_in[22];
  const float* cbn_g  = (const float*)d_in[23];
  const float* cbn_b  = (const float*)d_in[24];
  const float* pw2_w  = (const float*)d_in[25];
  const float* pw2_b  = (const float*)d_in[26];
  const float* ff2_g  = (const float*)d_in[27];
  const float* ff2_b  = (const float*)d_in[28];
  const float* ff2_w1 = (const float*)d_in[29];
  const float* ff2_b1 = (const float*)d_in[30];
  const float* ff2_w2 = (const float*)d_in[31];
  const float* ff2_b2 = (const float*)d_in[32];
  const float* fin_g  = (const float*)d_in[33];
  const float* fin_b  = (const float*)d_in[34];

  char* ws = (char*)d_ws;
  size_t off = 0;
  auto alloc = [&](size_t bytes) -> void* {
    void* p = ws + off;
    off += (bytes + 255) & ~(size_t)255;
    return p;
  };

  // bf16 weights
  unsigned short* w1b   = (unsigned short*)alloc((size_t)FF * DIM * 2);
  unsigned short* w2b   = (unsigned short*)alloc((size_t)DIM * FF * 2);
  unsigned short* wqb   = (unsigned short*)alloc((size_t)DIM * DIM * 2);
  unsigned short* wkb   = (unsigned short*)alloc((size_t)DIM * DIM * 2);
  unsigned short* wvb   = (unsigned short*)alloc((size_t)DIM * DIM * 2);
  unsigned short* wob   = (unsigned short*)alloc((size_t)DIM * DIM * 2);
  unsigned short* pw1b  = (unsigned short*)alloc((size_t)2 * CI * DIM * 2);
  unsigned short* pw2b  = (unsigned short*)alloc((size_t)DIM * CI * 2);
  unsigned short* f2w1b = (unsigned short*)alloc((size_t)FF * DIM * 2);
  unsigned short* f2w2b = (unsigned short*)alloc((size_t)DIM * FF * 2);
  unsigned short* dconvb= (unsigned short*)alloc((size_t)KW * CI * CI * 2);
  // activations
  unsigned short* xb    = (unsigned short*)alloc((size_t)B * DIM * S * 2);
  unsigned short* hbuf  = (unsigned short*)alloc((size_t)B * FF * S * 2);
  float* x1  = (float*)alloc((size_t)B * DIM * S * 4);
  float* x2  = (float*)alloc((size_t)B * DIM * S * 4);
  float* x3  = (float*)alloc((size_t)B * DIM * S * 4);
  float* x4  = (float*)alloc((size_t)B * DIM * S * 4);
  float* qf  = (float*)alloc((size_t)B * DIM * S * 4);
  float* vf  = (float*)alloc((size_t)B * DIM * S * 4);
  unsigned short* kbf = (unsigned short*)alloc((size_t)B * DIM * S * 2);
  unsigned short* qT  = (unsigned short*)alloc((size_t)B * H * S * DH * 2);
  unsigned short* vT  = (unsigned short*)alloc((size_t)B * H * S * DH * 2);
  float* scores = (float*)alloc((size_t)B * H * S * S * 4);
  unsigned short* Wb  = (unsigned short*)alloc((size_t)B * H * S * S * 2);
  float* oTf = (float*)alloc((size_t)B * H * S * DH * 4);
  unsigned short* obf = (unsigned short*)alloc((size_t)B * DIM * S * 2);
  float* big = (float*)alloc((size_t)B * 2 * CI * S * 4);
  unsigned short* gbuf = (unsigned short*)alloc((size_t)B * CI * S * 2);
  unsigned short* hb   = (unsigned short*)alloc((size_t)B * CI * S * 2);

  auto cvt = [&](const float* src, unsigned short* dst, long n) {
    k_f32_to_bf16<<<dim3((unsigned)((n + 255) / 256)), dim3(256), 0, stream>>>(src, dst, n);
  };
  auto bnb16 = [&](const float* xin, const float* g, const float* b, unsigned short* dst, int C) {
    long n = (long)B * C * S;
    k_bn_bf16<<<dim3((unsigned)((n + 255) / 256)), dim3(256), 0, stream>>>(xin, g, b, dst, n, C, S);
  };
  auto gemm = [&](bool conv,
                  const unsigned short* A, long sAz, int lda, int taps, long tapA, int pad,
                  const unsigned short* Bm, long sBz, int ldb, int Nsrc,
                  float* Cf, unsigned short* Cb, long sCz, int ldc,
                  const float* bias, const float* bng, const float* bnbp,
                  const float* resid, long sRz, float alpha,
                  int M, int N, int Kd, int act, int Z) {
    GemmP p;
    p.A = A; p.sAz = sAz; p.lda = lda; p.tapStrideA = tapA; p.taps = taps; p.pad = pad;
    p.Bm = Bm; p.sBz = sBz; p.ldb = ldb; p.Nsrc = Nsrc;
    p.Cf = Cf; p.Cb = Cb; p.sCz = sCz; p.ldc = ldc;
    p.bias = bias; p.bng = bng; p.bnb = bnbp;
    p.resid = resid; p.sRz = sRz;
    p.alpha = alpha; p.M = M; p.N = N; p.Kd = Kd; p.act = act;
    dim3 g((unsigned)((N + BN - 1) / BN), (unsigned)((M + BM - 1) / BM), (unsigned)Z);
    if (conv) k_gemm_wmma<true ><<<g, dim3(256), 0, stream>>>(p);
    else      k_gemm_wmma<false><<<g, dim3(256), 0, stream>>>(p);
  };

  // ---- weight conversions (deterministic, every call) ----
  cvt(ff1_w1, w1b,  (long)FF * DIM);
  cvt(ff1_w2, w2b,  (long)DIM * FF);
  cvt(wq, wqb, (long)DIM * DIM);
  cvt(wk, wkb, (long)DIM * DIM);
  cvt(wv, wvb, (long)DIM * DIM);
  cvt(wo, wob, (long)DIM * DIM);
  cvt(pw1_w, pw1b, (long)2 * CI * DIM);
  cvt(pw2_w, pw2b, (long)DIM * CI);
  cvt(ff2_w1, f2w1b, (long)FF * DIM);
  cvt(ff2_w2, f2w2b, (long)DIM * FF);
  {
    long n = (long)CI * CI * KW;
    k_reorder_dconv<<<dim3((unsigned)((n + 255) / 256)), dim3(256), 0, stream>>>(dconv_w, dconvb, CI, CI, KW);
  }

  const long actZ = (long)DIM * S;   // per-batch activation stride (512 ch)
  const long ffZ  = (long)FF * S;

  // ---- FF1: x1 = x + 0.5*W2(SiLU(W1 bn(x)+b1))+b2 ----
  bnb16(x, ff1_g, ff1_b, xb, DIM);
  gemm(false, w1b, 0, DIM, 1, 0, 0, xb, actZ, S, S, nullptr, hbuf, ffZ, S,
       ff1_b1, nullptr, nullptr, nullptr, 0, 1.0f, FF, S, DIM, /*silu*/1, B);
  gemm(false, w2b, 0, FF, 1, 0, 0, hbuf, ffZ, S, S, x1, nullptr, actZ, S,
       ff1_b2, nullptr, nullptr, x, actZ, 0.5f, DIM, S, FF, 0, B);

  // ---- Attention: x2 = x1 + Wo(softmax(qT k /8) vT) ----
  bnb16(x1, attn_g, attn_b, xb, DIM);
  gemm(false, wqb, 0, DIM, 1, 0, 0, xb, actZ, S, S, qf, nullptr, actZ, S,
       bq, nullptr, nullptr, nullptr, 0, 1.0f, DIM, S, DIM, 0, B);
  gemm(false, wkb, 0, DIM, 1, 0, 0, xb, actZ, S, S, nullptr, kbf, actZ, S,
       bk, nullptr, nullptr, nullptr, 0, 1.0f, DIM, S, DIM, 0, B);
  gemm(false, wvb, 0, DIM, 1, 0, 0, xb, actZ, S, S, vf, nullptr, actZ, S,
       bv, nullptr, nullptr, nullptr, 0, 1.0f, DIM, S, DIM, 0, B);
  {
    dim3 g((unsigned)(((long)DH * S + 255) / 256), 1, (unsigned)(B * H));
    k_transpose_cvt<<<g, dim3(256), 0, stream>>>(qf, qT, DH, S);   // [64,S] -> [S,64]
    k_transpose_cvt<<<g, dim3(256), 0, stream>>>(vf, vT, DH, S);
  }
  // scores[z] = 0.125 * qT[z] @ k[z]   (M=S, N=S, K=64), z = b*H+h
  gemm(false, qT, (long)S * DH, DH, 1, 0, 0, kbf, (long)DH * S, S, S,
       scores, nullptr, (long)S * S, S,
       nullptr, nullptr, nullptr, nullptr, 0, 0.125f, S, S, DH, 0, B * H);
  k_softmax_bf16<<<dim3((unsigned)(B * H * S)), dim3(128), 0, stream>>>(scores, Wb, S);
  // oT[z] = W[z] @ vT[z]   (M=S, N=64, K=S)
  gemm(false, Wb, (long)S * S, S, 1, 0, 0, vT, (long)S * DH, DH, DH,
       oTf, nullptr, (long)S * DH, DH,
       nullptr, nullptr, nullptr, nullptr, 0, 1.0f, S, DH, S, 0, B * H);
  {
    dim3 g((unsigned)(((long)S * DH + 255) / 256), 1, (unsigned)(B * H));
    k_transpose_cvt<<<g, dim3(256), 0, stream>>>(oTf, obf, S, DH); // [S,64] -> [64,S]
  }
  gemm(false, wob, 0, DIM, 1, 0, 0, obf, actZ, S, S, x2, nullptr, actZ, S,
       bo, nullptr, nullptr, x1, actZ, 1.0f, DIM, S, DIM, 0, B);

  // ---- Conv module: x3 = x2 + PW2(SiLU(bn(DenseConv(GLU(PW1 bn(x2)))))) ----
  bnb16(x2, conv_g, conv_b, xb, DIM);
  gemm(false, pw1b, 0, DIM, 1, 0, 0, xb, actZ, S, S, big, nullptr, (long)2 * CI * S, S,
       pw1_b, nullptr, nullptr, nullptr, 0, 1.0f, 2 * CI, S, DIM, 0, B);
  {
    long n = (long)B * CI * S;
    k_glu_bf16<<<dim3((unsigned)((n + 255) / 256)), dim3(256), 0, stream>>>(big, gbuf, CI, S, B);
  }
  // dense conv as 31-tap GEMM: M=1024, N=S, K=1024 per tap; fused bias+BN+SiLU -> bf16
  gemm(true, dconvb, 0, CI, KW, (long)CI * CI, PADC, gbuf, (long)CI * S, S, S,
       nullptr, hb, (long)CI * S, S,
       dconv_b, cbn_g, cbn_b, nullptr, 0, 1.0f, CI, S, CI, /*silu*/1, B);
  gemm(false, pw2b, 0, CI, 1, 0, 0, hb, (long)CI * S, S, S, x3, nullptr, actZ, S,
       pw2_b, nullptr, nullptr, x2, actZ, 1.0f, DIM, S, CI, 0, B);

  // ---- FF2 ----
  bnb16(x3, ff2_g, ff2_b, xb, DIM);
  gemm(false, f2w1b, 0, DIM, 1, 0, 0, xb, actZ, S, S, nullptr, hbuf, ffZ, S,
       ff2_b1, nullptr, nullptr, nullptr, 0, 1.0f, FF, S, DIM, /*silu*/1, B);
  gemm(false, f2w2b, 0, FF, 1, 0, 0, hbuf, ffZ, S, S, x4, nullptr, actZ, S,
       ff2_b2, nullptr, nullptr, x3, actZ, 0.5f, DIM, S, FF, 0, B);

  // ---- final BN -> d_out ----
  {
    long n = (long)B * DIM * S;
    k_bn_f32<<<dim3((unsigned)((n + 255) / 256)), dim3(256), 0, stream>>>(
        x4, fin_g, fin_b, (float*)d_out, n, DIM, S);
  }
}
